// GCN_3745211482328
// MI455X (gfx1250) — matrix operand order repaired and checked by reading, not compiled, not run on hardware
//
#include <hip/hip_runtime.h>

#define FEAT 128
#define LDSTRIDE 136   // halfwords; 272 B row stride -> 16B aligned, bank-spread

typedef __attribute__((ext_vector_type(16))) __bf16 v16bf;
typedef __attribute__((ext_vector_type(8)))  float  v8f;

__device__ __forceinline__ __bf16 f2bf(float f) { return (__bf16)f; }

// ---------------------------------------------------------------------------
// WMMA GEMM: H[nrows x 128] = (RELU? relu(X) : X) @ W   (W is 128x128, f32)
// Block = 256 threads = 8 waves; each wave owns a 16-row strip (128 rows/block).
// W is staged transposed in LDS as bf16 so each B fragment is 2x ds_load_b128.
// ---------------------------------------------------------------------------
template<bool RELU>
__global__ __launch_bounds__(256)
void gcn_gemm_wmma(const float* __restrict__ X,
                   const float* __restrict__ W,
                   float* __restrict__ H,
                   int nrows)
{
  __shared__ __bf16 Wl[FEAT * LDSTRIDE];

  const int t = threadIdx.x;
  // Stage W^T into LDS as bf16: Wl[n*LDSTRIDE + k] = bf16(W[k*128 + n])
  for (int idx = t; idx < FEAT * FEAT; idx += 256) {
    const int k = idx >> 7;
    const int n = idx & (FEAT - 1);
    Wl[n * LDSTRIDE + k] = f2bf(W[idx]);
  }
  __syncthreads();

  const int wave = t >> 5;
  const int lane = t & 31;
  const int hi   = lane >> 4;   // half-wave select (ISA 16-bit A/B layouts)
  const int lo   = lane & 15;

  const int row_base = blockIdx.x * 128 + wave * 16;
  int arow = row_base + lo;
  if (arow >= nrows) arow = nrows - 1;          // clamp tail loads
  const float4* Xr = (const float4*)(X + (size_t)arow * FEAT);

  v8f acc[8] = {};

  #pragma unroll
  for (int kb = 0; kb < 4; ++kb) {
    // A fragment (16x32 bf16): lane lo row; K runs [kb*32+hi*8,+8) and +16
    const int q0 = kb * 8 + hi * 2;             // float4 index of run0
    float4 x0 = Xr[q0];
    float4 x1 = Xr[q0 + 1];
    float4 x2 = Xr[q0 + 4];
    float4 x3 = Xr[q0 + 5];
    if (RELU) {
      x0.x = fmaxf(x0.x, 0.f); x0.y = fmaxf(x0.y, 0.f); x0.z = fmaxf(x0.z, 0.f); x0.w = fmaxf(x0.w, 0.f);
      x1.x = fmaxf(x1.x, 0.f); x1.y = fmaxf(x1.y, 0.f); x1.z = fmaxf(x1.z, 0.f); x1.w = fmaxf(x1.w, 0.f);
      x2.x = fmaxf(x2.x, 0.f); x2.y = fmaxf(x2.y, 0.f); x2.z = fmaxf(x2.z, 0.f); x2.w = fmaxf(x2.w, 0.f);
      x3.x = fmaxf(x3.x, 0.f); x3.y = fmaxf(x3.y, 0.f); x3.z = fmaxf(x3.z, 0.f); x3.w = fmaxf(x3.w, 0.f);
    }
    v16bf a;
    a[0]  = f2bf(x0.x); a[1]  = f2bf(x0.y); a[2]  = f2bf(x0.z); a[3]  = f2bf(x0.w);
    a[4]  = f2bf(x1.x); a[5]  = f2bf(x1.y); a[6]  = f2bf(x1.z); a[7]  = f2bf(x1.w);
    a[8]  = f2bf(x2.x); a[9]  = f2bf(x2.y); a[10] = f2bf(x2.z); a[11] = f2bf(x2.w);
    a[12] = f2bf(x3.x); a[13] = f2bf(x3.y); a[14] = f2bf(x3.z); a[15] = f2bf(x3.w);

    const int kbase = kb * 32 + hi * 8;         // K runs [kbase,+8) and [kbase+16,+8)
    #pragma unroll
    for (int nt = 0; nt < 8; ++nt) {
      const __bf16* bp = &Wl[(nt * 16 + lo) * LDSTRIDE + kbase];
      union { v16bf v; float4 q[2]; } bu;
      bu.q[0] = *(const float4*)(bp);           // ds_load_b128
      bu.q[1] = *(const float4*)(bp + 16);      // ds_load_b128
      acc[nt] = __builtin_amdgcn_wmma_f32_16x16x32_bf16(
          false, a, false, bu.v, (short)0, acc[nt], false, false);
    }
  }

  // C/D layout: lane -> column lo, VGPR v -> row hi*8+v
  #pragma unroll
  for (int nt = 0; nt < 8; ++nt) {
    #pragma unroll
    for (int v = 0; v < 8; ++v) {
      const int r = row_base + hi * 8 + v;
      if (r < nrows)
        H[(size_t)r * FEAT + nt * 16 + lo] = acc[nt][v];
    }
  }
}

// ---------------------------------------------------------------------------
__global__ void k_fill(float* __restrict__ p, float v, int n) {
  int i = blockIdx.x * 256 + threadIdx.x;
  if (i < n) p[i] = v;
}

__global__ void k_deg_count(const int* __restrict__ dst, float* __restrict__ deg, int E) {
  int e = blockIdx.x * 256 + threadIdx.x;
  if (e < E) atomicAdd(&deg[dst[e]], 1.0f);
}

__global__ void k_deg_finalize(float* __restrict__ deg, int n) {
  int i = blockIdx.x * 256 + threadIdx.x;
  if (i < n) deg[i] = rsqrtf(deg[i]);          // deg >= 1 (self loop)
}

// A[i][f] = H[i][f]*dinv[i]^2 + bias[f]   (self-loop term + bias)
__global__ void k_selfbias(const float* __restrict__ H, const float* __restrict__ dinv,
                           const float* __restrict__ bias, float* __restrict__ A, int n) {
  int i = blockIdx.x * 256 + threadIdx.x;
  if (i < n) {
    const int node = i >> 7;
    const int f    = i & (FEAT - 1);
    const float di = dinv[node];
    A[i] = H[i] * di * di + bias[f];
  }
}

// One wave per edge: coalesced float4 gather of H[src], 4 f32 atomics/lane
__global__ __launch_bounds__(256)
void k_scatter(const float* __restrict__ H, const float* __restrict__ dinv,
               const int* __restrict__ src, const int* __restrict__ dst,
               float* __restrict__ A, int E) {
  const int wave = (blockIdx.x * 256 + threadIdx.x) >> 5;
  const int lane = threadIdx.x & 31;
  if (wave >= E) return;
  const int s = src[wave];
  const int d = dst[wave];
  const float nrm = dinv[s] * dinv[d];
  const float4 h = ((const float4*)(H + (size_t)s * FEAT))[lane];
  float* out = A + (size_t)d * FEAT + lane * 4;
  atomicAdd(out + 0, h.x * nrm);
  atomicAdd(out + 1, h.y * nrm);
  atomicAdd(out + 2, h.z * nrm);
  atomicAdd(out + 3, h.w * nrm);
}

// One wave per node: ReLU + scatter into per-graph sums
__global__ __launch_bounds__(256)
void k_pool(const float* __restrict__ A, const int* __restrict__ batch,
            float* __restrict__ gsum, float* __restrict__ gcnt, int N) {
  const int wave = (blockIdx.x * 256 + threadIdx.x) >> 5;
  const int lane = threadIdx.x & 31;
  if (wave >= N) return;
  const int g = batch[wave];
  float4 h = ((const float4*)(A + (size_t)wave * FEAT))[lane];
  h.x = fmaxf(h.x, 0.f); h.y = fmaxf(h.y, 0.f);
  h.z = fmaxf(h.z, 0.f); h.w = fmaxf(h.w, 0.f);
  float* out = gsum + (size_t)g * FEAT + lane * 4;
  atomicAdd(out + 0, h.x);
  atomicAdd(out + 1, h.y);
  atomicAdd(out + 2, h.z);
  atomicAdd(out + 3, h.w);
  if (lane == 0) atomicAdd(&gcnt[g], 1.0f);
}

// out[g] = dot(gsum[g], linW)/max(cnt,1) + lin_b
__global__ void k_final(const float* __restrict__ gsum, const float* __restrict__ gcnt,
                        const float* __restrict__ linW, const float* __restrict__ linb,
                        float* __restrict__ out, int G) {
  int g = blockIdx.x * 256 + threadIdx.x;
  if (g >= G) return;
  float acc = 0.f;
  #pragma unroll 4
  for (int f = 0; f < FEAT; ++f) acc += gsum[(size_t)g * FEAT + f] * linW[f];
  float c = gcnt[g];
  c = c < 1.f ? 1.f : c;
  out[g] = acc / c + linb[0];
}

// ---------------------------------------------------------------------------
extern "C" void kernel_launch(void* const* d_in, const int* in_sizes, int n_in,
                              void* d_out, int out_size, void* d_ws, size_t ws_size,
                              hipStream_t stream) {
  const float* x     = (const float*)d_in[0];
  const int*   ei    = (const int*)  d_in[1];
  const int*   batch = (const int*)  d_in[2];
  const float* W1    = (const float*)d_in[3];
  const float* b1    = (const float*)d_in[4];
  const float* W2    = (const float*)d_in[5];
  const float* b2    = (const float*)d_in[6];
  const float* linW  = (const float*)d_in[7];
  const float* linb  = (const float*)d_in[8];
  float* out = (float*)d_out;

  const int N = in_sizes[0] / FEAT;
  const int E = in_sizes[1] / 2;
  const int G = out_size;
  const int* src = ei;
  const int* dst = ei + E;

  // workspace carve-out
  char* ws = (char*)d_ws;
  size_t off = 0;
  auto carve = [&](size_t bytes) -> char* {
    char* p = ws + off;
    off = (off + bytes + 255) & ~(size_t)255;
    return p;
  };
  float* H    = (float*)carve((size_t)N * FEAT * sizeof(float));
  float* A    = (float*)carve((size_t)N * FEAT * sizeof(float));
  float* dinv = (float*)carve((size_t)N * sizeof(float));
  float* gsum = (float*)carve((size_t)G * FEAT * sizeof(float));
  float* gcnt = (float*)carve((size_t)G * sizeof(float));

  const int nf        = N * FEAT;
  const int blkElems  = (nf + 255) / 256;
  const int blkNodes  = (N + 255) / 256;
  const int blkEdges  = (E + 255) / 256;
  const int blkGemm   = (N + 127) / 128;
  const int blkEWave  = (E + 7) / 8;          // wave per edge
  const int blkNWave  = (N + 7) / 8;          // wave per node

  // degrees (with self loop) -> dinv = rsqrt(deg)
  k_fill<<<blkNodes, 256, 0, stream>>>(dinv, 1.0f, N);
  k_deg_count<<<blkEdges, 256, 0, stream>>>(dst, dinv, E);
  k_deg_finalize<<<blkNodes, 256, 0, stream>>>(dinv, N);

  // layer 1: H = X @ W1 ; A = dinv^2*H + b1 ; A += scatter(H)
  gcn_gemm_wmma<false><<<blkGemm, 256, 0, stream>>>(x, W1, H, N);
  k_selfbias<<<blkElems, 256, 0, stream>>>(H, dinv, b1, A, nf);
  k_scatter<<<blkEWave, 256, 0, stream>>>(H, dinv, src, dst, A, E);

  // layer 2: H = relu(A) @ W2 ; A = dinv^2*H + b2 ; A += scatter(H)
  gcn_gemm_wmma<true><<<blkGemm, 256, 0, stream>>>(A, W2, H, N);
  k_selfbias<<<blkElems, 256, 0, stream>>>(H, dinv, b2, A, nf);
  k_scatter<<<blkEWave, 256, 0, stream>>>(H, dinv, src, dst, A, E);

  // global mean pool (ReLU fused) + final linear
  k_fill<<<(G * FEAT + 255) / 256, 256, 0, stream>>>(gsum, 0.0f, G * FEAT);
  k_fill<<<(G + 255) / 256, 256, 0, stream>>>(gcnt, 0.0f, G);
  k_pool<<<blkNWave, 256, 0, stream>>>(A, batch, gsum, gcnt, N);
  k_final<<<(G + 255) / 256, 256, 0, stream>>>(gsum, gcnt, linW, linb, out, G);
}
